// KLDivergenceProbLoss_44255343018047
// MI455X (gfx1250) — compile-verified, matrix operand order checked by reading, not run on hardware
//
#include <hip/hip_runtime.h>

typedef _Float16 v16h __attribute__((ext_vector_type(16)));
typedef float    v8f  __attribute__((ext_vector_type(8)));
typedef unsigned v8u  __attribute__((ext_vector_type(8)));

#define BATCH   16
#define NPER    (512*512)
#define NBINS   64
#define EPSF    1e-8f
#define WEIGHTF 0.1f
// sqrt( (1/(2*w^2)) * log2(e) ) with w = 1/64 : sqrt(2048 * 1.4426950408889634)
#define SQK     54.35659749f
#define LN2F    0.6931471805599453f

// order-preserving float<->uint keys for atomic min/max
__device__ __forceinline__ unsigned fkey(float f){
  unsigned u = __float_as_uint(f);
  return (u & 0x80000000u) ? ~u : (u | 0x80000000u);
}
__device__ __forceinline__ float fdec(unsigned u){
  return __uint_as_float((u & 0x80000000u) ? (u & 0x7fffffffu) : ~u);
}

// ---------------------------------------------------------------- init ws
__global__ __launch_bounds__(256) void init_ws(float* hist, unsigned* mm){
  int t = threadIdx.x;
  for (int i = t; i < 2*BATCH*NBINS; i += 256) hist[i] = 0.f;
  if (t < 2*BATCH) mm[t] = (t & 1) ? 0u : 0xFFFFFFFFu;   // [min-key, max-key] per batch
}

// ---------------------------------------------------------------- pass 1: per-batch min/max of target
__global__ __launch_bounds__(256) void minmax_k(const float4* __restrict__ targ, unsigned* mm){
  const int b = blockIdx.x, tid = threadIdx.x;
  const float4* tp = targ + (size_t)b * (NPER/4);
  float vmin =  __builtin_inff();
  float vmax = -__builtin_inff();
  const int stride = gridDim.y * blockDim.x;
  for (int i = blockIdx.y * blockDim.x + tid; i < NPER/4; i += stride){
    float4 v = tp[i];
    vmin = fminf(vmin, fminf(fminf(v.x, v.y), fminf(v.z, v.w)));
    vmax = fmaxf(vmax, fmaxf(fmaxf(v.x, v.y), fmaxf(v.z, v.w)));
  }
  __shared__ float smin[256], smax[256];
  smin[tid] = vmin; smax[tid] = vmax;
  __syncthreads();
  for (int s = 128; s > 0; s >>= 1){
    if (tid < s){
      smin[tid] = fminf(smin[tid], smin[tid+s]);
      smax[tid] = fmaxf(smax[tid], smax[tid+s]);
    }
    __syncthreads();
  }
  if (tid == 0){
    atomicMin(&mm[2*b+0], fkey(smin[0]));
    atomicMax(&mm[2*b+1], fkey(smax[0]));
  }
}

// ---------------------------------------------------------------- pass 2: WMMA-accumulated soft histograms
// async global->LDS double-buffered staging + WMMA(B=kern, A=ones) accumulation
__global__ __launch_bounds__(256) void hist_k(const float4* __restrict__ pred,
                                              const float4* __restrict__ targ,
                                              const unsigned* __restrict__ mm,
                                              float* __restrict__ histP,
                                              float* __restrict__ histT){
  const int b    = blockIdx.x;
  const int tid  = threadIdx.x;
  const int lane = tid & 31;
  const int half = lane >> 4;          // 0: lanes 0-15, 1: lanes 16-31
  const int col  = lane & 15;          // bin column within a 16-bin tile
  const int w    = tid >> 5;           // wave id in block
  const int wave   = blockIdx.y * (256/32) + w;
  const int nwaves = gridDim.y * (256/32);

  const float vmin = fdec(mm[2*b+0]);
  const float vmax = fdec(mm[2*b+1]);
  const float inv  = 1.0f / (vmax - vmin + EPSF);
  // d' = x*(inv*SQK) - (vmin*inv + c)*SQK ; kern = exp2(-(d'*d'))
  const float invS  = inv * SQK;
  const float baseS = vmin * inv * SQK;

  const float4* p4 = pred + (size_t)b * (NPER/4);
  const float4* t4 = targ + (size_t)b * (NPER/4);

  // per-wave private double buffer: [wave][buf][tensor][8 x float4 = 128B]
  __shared__ __align__(16) float4 lbuf[8][2][2][8];

  // per-lane LDS byte addresses for the async writes (lane stores 4B)
  unsigned ldsP[2], ldsT[2];
  #pragma unroll
  for (int i = 0; i < 2; ++i){
    ldsP[i] = (unsigned)(unsigned long long)(uintptr_t)&lbuf[w][i][0][0] + ((unsigned)lane << 2);
    ldsT[i] = (unsigned)(unsigned long long)(uintptr_t)&lbuf[w][i][1][0] + ((unsigned)lane << 2);
  }

  // one instruction moves a whole 128B group (wave32 x 4B), tracked by ASYNCcnt
  auto issue = [&](int grp, int bix){
    unsigned voff = (unsigned)grp * 128u + ((unsigned)lane << 2);
    asm volatile("global_load_async_to_lds_b32 %0, %1, %2"
                 :: "v"(ldsP[bix]), "v"(voff), "s"(p4) : "memory");
    asm volatile("global_load_async_to_lds_b32 %0, %1, %2"
                 :: "v"(ldsT[bix]), "v"(voff), "s"(t4) : "memory");
  };

  v16h ones;
  #pragma unroll
  for (int i = 0; i < 16; ++i) ones[i] = (_Float16)1.0f;

  v8f accP[4], accT[4];
  #pragma unroll
  for (int i = 0; i < 4; ++i){ accP[i] = (v8f)(0.f); accT[i] = (v8f)(0.f); }

  // kern evaluator: 16 x-values vs this lane's bin; B operand assembled from
  // whole 32-bit pkrtz results so each f16 pair lands verbatim in one VGPR.
  auto mk = [&](const float* xs, float ccS) -> v16h {
    v8u kb;
    #pragma unroll
    for (int i = 0; i < 8; ++i){
      float d0 = fmaf(xs[2*i+0], invS, -ccS);
      float d1 = fmaf(xs[2*i+1], invS, -ccS);
      float e0 = __builtin_amdgcn_exp2f(-(d0 * d0));   // v_mul v,v,-v ; v_exp_f32
      float e1 = __builtin_amdgcn_exp2f(-(d1 * d1));
      kb[i] = __builtin_bit_cast(unsigned, __builtin_amdgcn_cvt_pkrtz(e0, e1));
    }
    return __builtin_bit_cast(v16h, kb);
  };

  const int ngroups = NPER / 32;                        // 8192
  const int niter   = ngroups / nwaves;                 // 32, uniform, exact

  issue(wave, 0);
  int g = wave, cur = 0;
  for (int it = 0; it < niter; ++it){
    const int gn = g + nwaves;
    const int gl = (it + 1 < niter) ? gn : wave;        // last iter: dummy prefetch
    issue(gl, cur ^ 1);
    // allow the 2 just-issued to remain in flight; prior pair (current buf) done
    asm volatile("s_wait_asynccnt 2" ::: "memory");

    const float4* xpB = &lbuf[w][cur][0][half * 4];     // lanes share addrs -> LDS broadcast
    const float4* xtB = &lbuf[w][cur][1][half * 4];
    float4 a0 = xpB[0], a1 = xpB[1], a2 = xpB[2], a3 = xpB[3];
    float4 c0 = xtB[0], c1 = xtB[1], c2 = xtB[2], c3 = xtB[3];
    const float xsp[16] = {a0.x,a0.y,a0.z,a0.w, a1.x,a1.y,a1.z,a1.w,
                           a2.x,a2.y,a2.z,a2.w, a3.x,a3.y,a3.z,a3.w};
    const float xst[16] = {c0.x,c0.y,c0.z,c0.w, c1.x,c1.y,c1.z,c1.w,
                           c2.x,c2.y,c2.z,c2.w, c3.x,c3.y,c3.z,c3.w};

    #pragma unroll
    for (int tl = 0; tl < 4; ++tl){                     // four 16-bin tiles -> 64 bins
      const float ccS = baseS + ((float)(tl*16 + col) + 0.5f) * (SQK/64.0f);
      v16h kp = mk(xsp, ccS);
      accP[tl] = __builtin_amdgcn_wmma_f32_16x16x32_f16(false, ones, false, kp,
                                                        (short)0, accP[tl], false, false);
      v16h kt = mk(xst, ccS);
      accT[tl] = __builtin_amdgcn_wmma_f32_16x16x32_f16(false, ones, false, kt,
                                                        (short)0, accT[tl], false, false);
    }

    cur ^= 1;
    g = gn;
  }

  // every row of D equals the bin sums (A = ones); row 0 lives in C VGPR0, lanes 0-15
  if (lane < 16){
    #pragma unroll
    for (int tl = 0; tl < 4; ++tl){
      unsafeAtomicAdd(&histP[b*NBINS + tl*16 + lane], accP[tl][0]);
      unsafeAtomicAdd(&histT[b*NBINS + tl*16 + lane], accT[tl][0]);
    }
  }
}

// ---------------------------------------------------------------- pass 3: KL divergence + mean
__global__ __launch_bounds__(32) void finalize_k(const float* __restrict__ histP,
                                                 const float* __restrict__ histT,
                                                 float* __restrict__ out){
  const int t = threadIdx.x;
  float acc = 0.f;
  if (t < BATCH){
    const float* hp = histP + t*NBINS;
    const float* ht = histT + t*NBINS;
    float sp = 0.f, st = 0.f;
    for (int i = 0; i < NBINS; ++i){ sp += hp[i]; st += ht[i]; }
    const float isp = 1.f / (sp + EPSF);
    const float ist = 1.f / (st + EPSF);
    float kl = 0.f;
    for (int i = 0; i < NBINS; ++i){
      float p = hp[i] * isp;
      float q = ht[i] * ist;
      kl += q * ((__builtin_amdgcn_logf(q + EPSF) - __builtin_amdgcn_logf(p + EPSF)) * LN2F);
    }
    acc = kl;
  }
  #pragma unroll
  for (int off = 8; off > 0; off >>= 1) acc += __shfl_down(acc, off, 32);
  if (t == 0) out[0] = WEIGHTF * acc * (1.0f / (float)BATCH);
}

// ---------------------------------------------------------------- launch
extern "C" void kernel_launch(void* const* d_in, const int* in_sizes, int n_in,
                              void* d_out, int out_size, void* d_ws, size_t ws_size,
                              hipStream_t stream){
  const float* pred = (const float*)d_in[0];
  const float* targ = (const float*)d_in[1];
  float* out = (float*)d_out;

  float*    histP = (float*)d_ws;                      // 16*64 f32
  float*    histT = histP + BATCH*NBINS;               // 16*64 f32
  unsigned* mm    = (unsigned*)(histT + BATCH*NBINS);  // 16*2  u32

  init_ws<<<1, 256, 0, stream>>>(histP, mm);
  minmax_k<<<dim3(BATCH, 32), 256, 0, stream>>>((const float4*)targ, mm);
  hist_k<<<dim3(BATCH, 32), 256, 0, stream>>>((const float4*)pred, (const float4*)targ,
                                              mm, histP, histT);
  finalize_k<<<1, 32, 0, stream>>>(histP, histT, out);
}
